// RelativeMultiHeadAttention_80977313399661
// MI455X (gfx1250) — compile-verified, matrix operand order checked by reading, not run on hardware
//
#include <hip/hip_runtime.h>
#include <hip/hip_bf16.h>

// ---------------- problem constants ----------------
#define BATCH 8
#define CH    384
#define SEQ   1024
#define NH    6
#define HD    64
#define BQ    32      // query rows per attention workgroup

typedef __attribute__((ext_vector_type(16))) _Float16 v16h;
typedef __attribute__((ext_vector_type(8)))  float    v8f;

// ---------------------------------------------------------------------------
// WMMA 16x16x32 f16 fragment helpers (CDNA5 ISA 7.12.2 layouts, wave32).
// A (16x32, 16-bit): lane group [0,16) holds K offsets {0..7}U{16..23},
// lanes [16,32) hold {8..15}U{24..31}, packed as consecutive pairs.
// B mirrors A with lane -> column. C/D: 8 f32, row M = r + 8*(lane>>4),
// col N = lane & 15.
// ---------------------------------------------------------------------------
__device__ __forceinline__ int frag_k(int pr, int laneHi) {
  int kb = laneHi ? 8 : 0;
  return (pr < 4) ? (kb + 2 * pr) : (16 + kb + 2 * (pr - 4));
}

// K contiguous in memory: element (idx, k) at p[idx*ld + k]  (A rows / B cols)
__device__ __forceinline__ v16h load_frag_kc(const _Float16* p, int idx0, int ld, int k0) {
  int lane = threadIdx.x & 31;
  int hi = lane >> 4;
  const _Float16* base = p + (size_t)(idx0 + (lane & 15)) * ld + k0;
  v16h f;
#pragma unroll
  for (int pr = 0; pr < 8; ++pr) {
    int k = frag_k(pr, hi);
    f[2 * pr]     = base[k];
    f[2 * pr + 1] = base[k + 1];
  }
  return f;
}

// A fragment sourced from f32 LDS (probabilities), converted to f16 inline
__device__ __forceinline__ v16h load_frag_lds(const float* s, int row0, int ld, int k0) {
  int lane = threadIdx.x & 31;
  int hi = lane >> 4;
  const float* base = s + (size_t)(row0 + (lane & 15)) * ld + k0;
  v16h f;
#pragma unroll
  for (int pr = 0; pr < 8; ++pr) {
    int k = frag_k(pr, hi);
    f[2 * pr]     = (_Float16)base[k];
    f[2 * pr + 1] = (_Float16)base[k + 1];
  }
  return f;
}

#define WMMA_F16(a, b, c) \
  __builtin_amdgcn_wmma_f32_16x16x32_f16(false, (a), false, (b), (short)0, (c), false, false)

// ---------------------------------------------------------------------------
// conversion kernels
// ---------------------------------------------------------------------------
__global__ void f32_to_f16_kernel(const float* __restrict__ s, _Float16* __restrict__ d, int n) {
  int i = blockIdx.x * 256 + threadIdx.x;
  if (i < n) d[i] = (_Float16)s[i];
}

// x (B,C,L) f32 -> xh (B,L,C) f16   (makes K = channels contiguous)
__global__ void transpose_x_kernel(const float* __restrict__ x, _Float16* __restrict__ xh) {
  int i = blockIdx.x * 256 + threadIdx.x;
  if (i >= BATCH * CH * SEQ) return;
  int l = i % SEQ;
  int c = (i / SEQ) % CH;
  int b = i / (SEQ * CH);
  xh[((size_t)b * SEQ + l) * CH + c] = (_Float16)x[i];
}

// ---------------------------------------------------------------------------
// QKV projection GEMM: Out[o,l] = sum_c W[o,c] * Xh[l,c] + bias, times qscale.
// One 16x16 output tile per wave, K loop in steps of 32 -> v_wmma per step.
// Optional dstT stores the per-head TRANSPOSED copy (d-major, seq contiguous)
// so the later P@V GEMM gets K-contiguous B fragments (b128 loads, no stride).
// ---------------------------------------------------------------------------
__global__ __launch_bounds__(256) void qkv_gemm_kernel(
    const _Float16* __restrict__ w16, const _Float16* __restrict__ xh,
    const float* __restrict__ bias, _Float16* __restrict__ dst,
    _Float16* __restrict__ dstT, float qscale) {
  int wave = threadIdx.x >> 5, lane = threadIdx.x & 31;
  int tile = blockIdx.x * 8 + wave;
  int tm = tile % (CH / 16);                         // 24 row tiles (channels out)
  int tn = tile / (CH / 16);                         // 64 col tiles (sequence)
  int b = blockIdx.y;
  int o0 = tm * 16, l0 = tn * 16;
  const _Float16* xb = xh + (size_t)b * SEQ * CH;

  v8f acc = {};
#pragma unroll 4
  for (int c0 = 0; c0 < CH; c0 += 32) {
    __builtin_prefetch(xb + (size_t)l0 * CH + c0 + 64, 0, 1);
    v16h a = load_frag_kc(w16, o0, CH, c0);
    v16h bb = load_frag_kc(xb, l0, CH, c0);
    acc = WMMA_F16(a, bb, acc);
  }

  int col = lane & 15, rbase = (lane >> 4) * 8;
#pragma unroll
  for (int r = 0; r < 8; ++r) {
    int o = o0 + rbase + r;
    float v = (acc[r] + bias[o]) * qscale;           // fold 1/sqrt(d) into q
    int h = o >> 6, dd = o & 63;
    int lg = l0 + col;
    _Float16 hv = (_Float16)v;
    dst[(((size_t)b * NH + h) * SEQ + lg) * HD + dd] = hv;
    if (dstT)
      dstT[(((size_t)b * NH + h) * HD + dd) * SEQ + lg] = hv;
  }
}

__global__ __launch_bounds__(256) void out_gemm_kernel(
    const _Float16* __restrict__ wo16, const _Float16* __restrict__ resh,
    const float* __restrict__ bo, float* __restrict__ out) {
  int wave = threadIdx.x >> 5, lane = threadIdx.x & 31;
  int tile = blockIdx.x * 8 + wave;
  int tm = tile % (CH / 16);
  int tn = tile / (CH / 16);
  int b = blockIdx.y;
  int o0 = tm * 16, l0 = tn * 16;
  const _Float16* rb = resh + (size_t)b * SEQ * CH;

  v8f acc = {};
#pragma unroll 4
  for (int c0 = 0; c0 < CH; c0 += 32) {
    v16h a = load_frag_kc(wo16, o0, CH, c0);
    v16h bb = load_frag_kc(rb, l0, CH, c0);
    acc = WMMA_F16(a, bb, acc);
  }

  int col = lane & 15, rbase = (lane >> 4) * 8;
#pragma unroll
  for (int r = 0; r < 8; ++r) {
    int o = o0 + rbase + r;
    out[((size_t)b * CH + o) * SEQ + l0 + col] = acc[r] + bo[o];
  }
}

// ---------------------------------------------------------------------------
// rq[b,h,l,off] = q'(scaled) . emb_rel_k[h,off,:]   (band of width 9)
// ---------------------------------------------------------------------------
__global__ void relk_dots_kernel(const _Float16* __restrict__ q,
                                 const float* __restrict__ embk,
                                 float* __restrict__ rq) {
  int i = blockIdx.x * 256 + threadIdx.x;
  if (i >= BATCH * NH * SEQ * 9) return;
  int off = i % 9;
  int l = (i / 9) % SEQ;
  int bh = i / (9 * SEQ);
  int h = bh % NH;
  const _Float16* qp = q + ((size_t)bh * SEQ + l) * HD;
  const float* e = embk + ((size_t)h * 9 + off) * HD;
  float s = 0.f;
#pragma unroll
  for (int dd = 0; dd < HD; ++dd) s += (float)qp[dd] * e[dd];
  rq[i] = s;
}

// ---------------------------------------------------------------------------
// Fused attention per (b,h, 32 query rows). 256 threads = 8 waves.
// LDS: 32 x 1024 f32 score strip (128 KB of the WGP's 320 KB).
//   stage 1 : S = Q K^T (WMMA) -> LDS
//   stage 1b: add banded relative-K scores
//   stage 2 : exact softmax per row (wave32 shuffle reductions)
//   stage 3 : O = P V (WMMA, P from LDS, V^T K-contiguous) + emb_rel_v band
// ---------------------------------------------------------------------------
__global__ __launch_bounds__(256) void attn_kernel(
    const _Float16* __restrict__ q, const _Float16* __restrict__ k,
    const _Float16* __restrict__ vT, const float* __restrict__ rq,
    const float* __restrict__ embv, _Float16* __restrict__ res) {
  extern __shared__ float sc[];                      // [BQ][SEQ]
  int bh = blockIdx.y;
  int b = bh / NH, h = bh % NH;
  int l0 = blockIdx.x * BQ;
  const _Float16* qh = q + (size_t)bh * SEQ * HD;
  const _Float16* kh = k + (size_t)bh * SEQ * HD;
  const _Float16* vhT = vT + (size_t)bh * HD * SEQ;  // [HD][SEQ], seq contiguous
  int tid = threadIdx.x, wave = tid >> 5, lane = tid & 31;
  int col = lane & 15, rbase = (lane >> 4) * 8;

  // ---- stage 1: scores -> LDS -------------------------------------------
  {
    int qt = wave & 1;                               // which 16-row qtile
    int ql = l0 + qt * 16;
    v16h a0 = load_frag_kc(qh, ql, HD, 0);
    v16h a1 = load_frag_kc(qh, ql, HD, 32);
    int g = wave >> 1;                               // 4 groups of 16 key tiles
#pragma unroll 2
    for (int t = 0; t < 16; ++t) {
      int j0 = (g * 16 + t) * 16;
      v16h b0 = load_frag_kc(kh, j0, HD, 0);
      v16h b1 = load_frag_kc(kh, j0, HD, 32);
      v8f acc = {};
      acc = WMMA_F16(a0, b0, acc);
      acc = WMMA_F16(a1, b1, acc);
#pragma unroll
      for (int r = 0; r < 8; ++r)
        sc[(qt * 16 + rbase + r) * SEQ + j0 + col] = acc[r];
    }
  }
  __syncthreads();

  // ---- stage 1b: banded relative-K term ---------------------------------
  for (int e = tid; e < BQ * 9; e += 256) {
    int lr = e / 9, off = e % 9;
    int lg = l0 + lr;
    int j = lg + off - 4;
    if (j >= 0 && j < SEQ)
      sc[lr * SEQ + j] += rq[((size_t)bh * SEQ + lg) * 9 + off];
  }
  __syncthreads();

  // ---- stage 2: softmax (4 rows per wave, exact two-pass) ---------------
  for (int rr = 0; rr < 4; ++rr) {
    float* row = sc + (wave * 4 + rr) * SEQ;
    float m = -1e30f;
    for (int j = lane; j < SEQ; j += 32) m = fmaxf(m, row[j]);
#pragma unroll
    for (int o = 16; o; o >>= 1) m = fmaxf(m, __shfl_xor(m, o, 32));
    float s = 0.f;
    for (int j = lane; j < SEQ; j += 32) {
      float e = __expf(row[j] - m);
      row[j] = e;
      s += e;
    }
#pragma unroll
    for (int o = 16; o; o >>= 1) s += __shfl_xor(s, o, 32);
    float inv = 1.0f / s;
    for (int j = lane; j < SEQ; j += 32) row[j] *= inv;
  }
  __syncthreads();

  // ---- stage 3: O = P V + band(P, emb_rel_v) ----------------------------
  {
    int qt = wave >> 2;                              // 2 qtiles
    int dt = wave & 3;                               // 4 d tiles of 16
    v8f acc = {};
    for (int j0 = 0; j0 < SEQ; j0 += 32) {
      v16h a = load_frag_lds(sc, qt * 16, SEQ, j0);
      v16h bb = load_frag_kc(vhT, dt * 16, SEQ, j0);  // K(=j) contiguous now
      acc = WMMA_F16(a, bb, acc);
    }
#pragma unroll
    for (int r = 0; r < 8; ++r) {
      int lr = qt * 16 + rbase + r;
      int lg = l0 + lr;
      int dd = dt * 16 + col;
      float val = acc[r];
#pragma unroll
      for (int off = 0; off < 9; ++off) {
        int j = lg + off - 4;
        if (j >= 0 && j < SEQ)
          val += sc[lr * SEQ + j] * embv[((size_t)h * 9 + off) * HD + dd];
      }
      res[((size_t)b * SEQ + lg) * CH + h * HD + dd] = (_Float16)val;
    }
  }
}

// ---------------------------------------------------------------------------
extern "C" void kernel_launch(void* const* d_in, const int* in_sizes, int n_in,
                              void* d_out, int out_size, void* d_ws, size_t ws_size,
                              hipStream_t stream) {
  const float* x   = (const float*)d_in[0];
  const float* wq  = (const float*)d_in[1];
  const float* bq  = (const float*)d_in[2];
  const float* wk  = (const float*)d_in[3];
  const float* bk  = (const float*)d_in[4];
  const float* wv  = (const float*)d_in[5];
  const float* bv  = (const float*)d_in[6];
  const float* wo  = (const float*)d_in[7];
  const float* bo  = (const float*)d_in[8];
  const float* embk = (const float*)d_in[9];
  const float* embv = (const float*)d_in[10];
  float* out = (float*)d_out;

  // workspace carve-up
  char* ws = (char*)d_ws;
  auto carve = [&](size_t bytes) { char* p = ws; ws += (bytes + 255) & ~(size_t)255; return p; };
  _Float16* wq16 = (_Float16*)carve((size_t)CH * CH * 2);
  _Float16* wk16 = (_Float16*)carve((size_t)CH * CH * 2);
  _Float16* wv16 = (_Float16*)carve((size_t)CH * CH * 2);
  _Float16* wo16 = (_Float16*)carve((size_t)CH * CH * 2);
  _Float16* xh   = (_Float16*)carve((size_t)BATCH * SEQ * CH * 2);
  _Float16* q16  = (_Float16*)carve((size_t)BATCH * NH * SEQ * HD * 2);
  _Float16* k16  = (_Float16*)carve((size_t)BATCH * NH * SEQ * HD * 2);
  _Float16* v16  = (_Float16*)carve((size_t)BATCH * NH * SEQ * HD * 2);
  _Float16* vT16 = (_Float16*)carve((size_t)BATCH * NH * HD * SEQ * 2);
  float*    rq   = (float*)carve((size_t)BATCH * NH * SEQ * 9 * 4);
  _Float16* res16 = (_Float16*)carve((size_t)BATCH * SEQ * CH * 2);

  const float scale = 0.125f;  // 1/sqrt(64)

  // 1) weight / activation conversion
  {
    int n = CH * CH, g = (n + 255) / 256;
    f32_to_f16_kernel<<<g, 256, 0, stream>>>(wq, wq16, n);
    f32_to_f16_kernel<<<g, 256, 0, stream>>>(wk, wk16, n);
    f32_to_f16_kernel<<<g, 256, 0, stream>>>(wv, wv16, n);
    f32_to_f16_kernel<<<g, 256, 0, stream>>>(wo, wo16, n);
  }
  {
    int n = BATCH * CH * SEQ;
    transpose_x_kernel<<<(n + 255) / 256, 256, 0, stream>>>(x, xh);
  }

  // 2) QKV projections (WMMA); scale folded into q; V also stored transposed
  {
    dim3 grid((CH / 16) * (SEQ / 16) / 8, BATCH);    // 192 x 8
    qkv_gemm_kernel<<<grid, 256, 0, stream>>>(wq16, xh, bq, q16, ((_Float16*)0), scale);
    qkv_gemm_kernel<<<grid, 256, 0, stream>>>(wk16, xh, bk, k16, ((_Float16*)0), 1.0f);
    qkv_gemm_kernel<<<grid, 256, 0, stream>>>(wv16, xh, bv, v16, vT16, 1.0f);
  }

  // 3) banded relative-K dot products
  {
    int n = BATCH * NH * SEQ * 9;
    relk_dots_kernel<<<(n + 255) / 256, 256, 0, stream>>>(q16, embk, rq);
  }

  // 4) fused attention (128 KB LDS score strip per workgroup)
  {
    dim3 grid(SEQ / BQ, BATCH * NH);                 // 32 x 48
    size_t lds = (size_t)BQ * SEQ * sizeof(float);   // 131072 B
    attn_kernel<<<grid, 256, lds, stream>>>(q16, k16, vT16, rq, embv, res16);
  }

  // 5) output projection (WMMA) -> f32 (B,C,L)
  {
    dim3 grid((CH / 16) * (SEQ / 16) / 8, BATCH);
    out_gemm_kernel<<<grid, 256, 0, stream>>>(wo16, res16, bo, out);
  }
}
